// RWKV7TimeMixer_41635412967453
// MI455X (gfx1250) — compile-verified, matrix operand order checked by reading, not run on hardware
//
#include <hip/hip_runtime.h>

#define DEVI __device__ __forceinline__

typedef unsigned short u16;
typedef __attribute__((ext_vector_type(16))) __bf16 v16bf;
typedef __attribute__((ext_vector_type(8)))  __bf16 v8bf;
typedef __attribute__((ext_vector_type(8)))  float  v8f;
typedef __attribute__((ext_vector_type(4)))  unsigned int u32x4;
typedef __attribute__((ext_vector_type(8)))  int i32x8;
typedef __attribute__((ext_vector_type(4)))  int i32x4;

constexpr int Bz = 8, Tz = 2048, Cz = 1024, Hz = 16, Nz = 64;
constexpr int BT = Bz * Tz;
constexpr float EPS_GN = 0.00064f;

// ---------- helpers ----------
DEVI u16 f2bf(float f) {
  unsigned u = __float_as_uint(f);
  u += 0x7FFFu + ((u >> 16) & 1u);     // round-to-nearest-even
  return (u16)(u >> 16);
}

// 16x32 bf16 fragment from row-major source (leading dim ld elements).
// CDNA5 16-bit A layout: lane L holds row L%16; lanes 0-15 cover
// K in [0,8)+[16,24), lanes 16-31 cover K in [8,16)+[24,32).
DEVI v16bf load_frag_rm(const u16* base, int ld, int lane) {
  int row = lane & 15;
  int kb  = (lane < 16) ? 0 : 8;
  const u16* p = base + (size_t)row * ld + kb;
  v8bf lo = *reinterpret_cast<const v8bf*>(p);
  v8bf hi = *reinterpret_cast<const v8bf*>(p + 16);
  v16bf r;
#pragma unroll
  for (int q = 0; q < 8; q++) { r[q] = lo[q]; r[8 + q] = hi[q]; }
  return r;
}

// Issue a TDM 2D tile load: rows x cols (bf16) from row-major global
// (stride = srcLd elements) into LDS at lds_off (packed rows x cols).
// D# layout per CDNA5 ISA ch.8 (group0: flags/lds/global/type, group1: dims).
DEVI void tdm_load_2d(unsigned lds_off, const void* gp,
                      unsigned rows, unsigned cols, unsigned srcLd) {
  unsigned long long ga = (unsigned long long)gp;
  u32x4 g0;
  g0[0] = 1u;                                        // count=1, no gather
  g0[1] = lds_off;                                   // LDS byte address
  g0[2] = (unsigned)(ga & 0xFFFFFFFFu);              // global_addr[31:0]
  g0[3] = (unsigned)((ga >> 32) & 0x1FFFFFFu) | (2u << 30);  // [56:32] | type=2
  i32x8 g1;
  g1[0] = (int)(1u << 16);                           // data_size=1 (2 bytes)
  g1[1] = (int)((cols & 0xFFFFu) << 16);             // tensor_dim0[15:0]
  g1[2] = (int)(((cols >> 16) & 0xFFFFu) | ((rows & 0xFFFFu) << 16)); // td0 hi | td1 lo
  g1[3] = (int)(((rows >> 16) & 0xFFFFu) | ((cols & 0xFFFFu) << 16)); // td1 hi | tile_dim0
  g1[4] = (int)(rows & 0xFFFFu);                     // tile_dim1 (tile_dim2=0)
  g1[5] = (int)srcLd;                                // tensor_dim0_stride[31:0]
  g1[6] = 0;                                         // stride hi | dim1_stride lo
  g1[7] = 0;
  i32x4 gz4 = {0, 0, 0, 0};
  i32x8 gz8 = {0, 0, 0, 0, 0, 0, 0, 0};
  __builtin_amdgcn_tensor_load_to_lds(g0, g1, gz4, gz4, gz8, 0);
}

// ---------- kernels ----------
__global__ void k_f2bf(const float* s, u16* d, int n) {
  int i = blockIdx.x * 256 + threadIdx.x;
  if (i < n) d[i] = f2bf(s[i]);
}

// convert + transpose: src [R,Cc] fp32 row-major -> dst [Cc,R] bf16 row-major
__global__ void k_f2bfT(const float* s, u16* d, int R_, int Cc) {
  int i = blockIdx.x * 256 + threadIdx.x;
  if (i >= R_ * Cc) return;
  int r = i / Cc, c = i % Cc;
  d[(size_t)c * R_ + r] = f2bf(s[i]);
}

__global__ void k_mix(const float* __restrict__ x,
                      const float* cr, const float* cw, const float* ck,
                      const float* cv, const float* ca, const float* cg,
                      u16* xr, u16* xw, u16* xk, u16* xv, u16* xa, u16* xg) {
  int i = blockIdx.x * 256 + threadIdx.x;        // < BT*C
  int c = i & (Cz - 1);
  int m = i >> 10;
  int t = m & (Tz - 1);
  float x0 = x[i];
  float pv = (t == 0) ? 0.f : x[i - Cz];
  float xx = pv - x0;
  xr[i] = f2bf(x0 + xx * cr[c]);
  xw[i] = f2bf(x0 + xx * cw[c]);
  xk[i] = f2bf(x0 + xx * ck[c]);
  xv[i] = f2bf(x0 + xx * cv[c]);
  xa[i] = f2bf(x0 + xx * ca[c]);
  xg[i] = f2bf(x0 + xx * cg[c]);
}

__global__ void k_act(const float* s, u16* d, int n, int mode) {
  int i = blockIdx.x * 256 + threadIdx.x;
  if (i >= n) return;
  float v = s[i];
  if (mode == 1) v = tanhf(v);
  else if (mode == 2) v = 1.f / (1.f + expf(-v));
  d[i] = f2bf(v);
}

// C[M,N] fp32 = A[M,K] bf16 @ B[N,K]^T  (B always stored [N,K] row-major).
// Block = 256 threads = 8 waves = 64x32 macro-tile; TDM double-buffers
// A (64x32) and B (32x32) tiles into LDS; WMMA feeds from LDS.
__global__ void gemm_bf16_wmma(const u16* __restrict__ A, const u16* __restrict__ Bm,
                               float* __restrict__ C, int M, int N, int K) {
  __shared__ u16 shA[2][64 * 32];
  __shared__ u16 shB[2][32 * 32];
  int wave = threadIdx.x >> 5;
  int lane = threadIdx.x & 31;
  int mBase = blockIdx.y * 64;
  int nBase = blockIdx.x * 32;
  int wm = wave & 3, wn = wave >> 2;

  auto issue = [&](int buf, int k0) {
    tdm_load_2d((unsigned)(size_t)(const void*)&shA[buf][0],
                A + (size_t)mBase * K + k0, 64, 32, (unsigned)K);
    tdm_load_2d((unsigned)(size_t)(const void*)&shB[buf][0],
                Bm + (size_t)nBase * K + k0, 32, 32, (unsigned)K);
  };

  if (wave == 0) issue(0, 0);

  v8f acc = {0.f, 0.f, 0.f, 0.f, 0.f, 0.f, 0.f, 0.f};
  int buf = 0;
  for (int k0 = 0; k0 < K; k0 += 32) {
    if (wave == 0) {
      if (k0 + 32 < K) {
        issue(buf ^ 1, k0 + 32);                 // prefetch next stage via TDM
        __builtin_amdgcn_s_wait_tensorcnt(2);    // current stage's 2 loads done
      } else {
        __builtin_amdgcn_s_wait_tensorcnt(0);
      }
    }
    __syncthreads();
    v16bf af  = load_frag_rm(&shA[buf][wm * 16 * 32], 32, lane);
    v16bf bfm = load_frag_rm(&shB[buf][wn * 16 * 32], 32, lane);
    acc = __builtin_amdgcn_wmma_f32_16x16x32_bf16(false, af, false, bfm,
                                                  (short)0, acc, false, false);
    __syncthreads();
    buf ^= 1;
  }

  int rbase = (lane < 16) ? 0 : 8;
  int col = nBase + wn * 16 + (lane & 15);
  int m0 = mBase + wm * 16;
#pragma unroll
  for (int q = 0; q < 8; q++)
    C[(size_t)(m0 + rbase + q) * N + col] = acc[q];
}

__global__ void k_prep(const float* r32, const float* k32, const float* v32,
                       const float* dw, const float* da, const float* dv,
                       const float* __restrict__ vfirst,
                       const float* w0, const float* a0, const float* v0,
                       const float* kkc, const float* kac,
                       float* R, float* W, float* K, float* V, float* A, float* Bb) {
  int i = blockIdx.x * 256 + threadIdx.x;        // BT*C
  int tid = threadIdx.x;
  int c = i & (Cz - 1);
  int m = i >> 10;
  int b = m >> 11;
  int t = m & (Tz - 1);
  int h = c >> 6;
  int j = c & 63;
  float kv = k32[i];
  float kk = kv * kkc[c];
  __shared__ float red[256];
  red[tid] = kk * kk;
  __syncthreads();
  int hb = tid & ~63;
  float ss = 0.f;
#pragma unroll
  for (int q = 0; q < 64; q++) ss += red[hb + q];
  float kkn = kk / fmaxf(sqrtf(ss), 1e-12f);

  float wr = w0[c] + dw[i];
  float wv = -log1pf(expf(-wr)) - 0.5f;          // -softplus(-x) - 0.5
  float decay = expf(-expf(wv));
  float av = 1.f / (1.f + expf(-(a0[c] + da[i])));
  float mx = 1.f / (1.f + expf(-(v0[c] + dv[i])));
  float vv = v32[i];
  float vf = vv + (vfirst[i] - vv) * mx;
  float kf = kv * (1.f + (av - 1.f) * kac[c]);

  size_t si = (((size_t)(b * Hz + h)) * Tz + t) * Nz + j;   // [B,H,T,N]
  R[si] = r32[i]; W[si] = decay; K[si] = kf;
  V[si] = vf;     A[si] = -kkn;  Bb[si] = kkn * av;
}

// One block per (b,h); 256 threads: thread = (row i, j-group jg of 16 cols).
// State s[64][64] lives in VGPRs (16 floats/thread).
__global__ void k_scan(const float* __restrict__ R, const float* __restrict__ W,
                       const float* __restrict__ K, const float* __restrict__ V,
                       const float* __restrict__ A, const float* __restrict__ Bb,
                       float* __restrict__ Y) {
  int bh = blockIdx.x;
  int tid = threadIdx.x;
  int i = tid >> 2, jg = tid & 3, j0 = jg << 4;
  size_t hoff = (size_t)bh * Tz * Nz;
  __shared__ float sh[6 * Nz];
  float s[16];
#pragma unroll
  for (int q = 0; q < 16; q++) s[q] = 0.f;

  for (int t = 0; t < Tz; t++) {
    size_t off = hoff + (size_t)t * Nz;
    {
      int li = tid;
      if (li < 6 * Nz) {
        int a = li >> 6, e = li & 63;
        const float* bp = a == 0 ? R : a == 1 ? W : a == 2 ? K
                        : a == 3 ? V : a == 4 ? A : Bb;
        sh[li] = bp[off + e];
      }
      li = tid + 256;
      if (li < 6 * Nz) {
        int a = li >> 6, e = li & 63;
        const float* bp = a == 0 ? R : a == 1 ? W : a == 2 ? K
                        : a == 3 ? V : a == 4 ? A : Bb;
        sh[li] = bp[off + e];
      }
    }
    __syncthreads();

    float sa = 0.f;
#pragma unroll
    for (int q = 0; q < 16; q++) sa += s[q] * sh[4 * Nz + j0 + q];
    sa += __shfl_xor(sa, 1);
    sa += __shfl_xor(sa, 2);

    float vi = sh[3 * Nz + i];
    float yp = 0.f;
#pragma unroll
    for (int q = 0; q < 16; q++) {
      int j = j0 + q;
      s[q] = s[q] * sh[Nz + j] + sa * sh[5 * Nz + j] + vi * sh[2 * Nz + j];
      yp += s[q] * sh[j];
    }
    yp += __shfl_xor(yp, 1);
    yp += __shfl_xor(yp, 2);
    if (jg == 0) Y[off + i] = yp;
    __syncthreads();
  }
}

__global__ void k_post(const float* __restrict__ Y, const float* R, const float* K,
                       const float* V, const float* g32,
                       const float* lng, const float* lnb, const float* rkc,
                       u16* __restrict__ zo) {
  int i = blockIdx.x * 256 + threadIdx.x;
  int tid = threadIdx.x;
  int c = i & (Cz - 1);
  int m = i >> 10;
  int b = m >> 11;
  int t = m & (Tz - 1);
  int h = c >> 6;
  int j = c & 63;
  size_t si = (((size_t)(b * Hz + h)) * Tz + t) * Nz + j;
  float y = Y[si];
  __shared__ float r1[256], r2[256];
  r1[tid] = y; r2[tid] = y * y;
  __syncthreads();
  int hb = tid & ~63;
  float su = 0.f, sq = 0.f;
#pragma unroll
  for (int q = 0; q < 64; q++) { su += r1[hb + q]; sq += r2[hb + q]; }
  float mu  = su * (1.f / 64.f);
  float var = sq * (1.f / 64.f) - mu * mu;
  float yn  = (y - mu) * rsqrtf(var + EPS_GN);
  float yl  = yn * lng[c] + lnb[c];
  __syncthreads();
  r1[tid] = R[si] * K[si] * rkc[c];
  __syncthreads();
  float rk = 0.f;
#pragma unroll
  for (int q = 0; q < 64; q++) rk += r1[hb + q];
  float z = (yl + rk * V[si]) * g32[i];
  zo[i] = f2bf(z);
}

// ---------- launch ----------
extern "C" void kernel_launch(void* const* d_in, const int* in_sizes, int n_in,
                              void* d_out, int out_size, void* d_ws, size_t ws_size,
                              hipStream_t stream) {
  (void)in_sizes; (void)n_in; (void)out_size; (void)ws_size;
  const float* x      = (const float*)d_in[0];
  const float* vfirst = (const float*)d_in[1];
  const float* x_r = (const float*)d_in[2];
  const float* x_w = (const float*)d_in[3];
  const float* x_k = (const float*)d_in[4];
  const float* x_v = (const float*)d_in[5];
  const float* x_a = (const float*)d_in[6];
  const float* x_g = (const float*)d_in[7];
  const float* w0  = (const float*)d_in[8];
  const float* w1  = (const float*)d_in[9];
  const float* w2  = (const float*)d_in[10];
  const float* a0  = (const float*)d_in[11];
  const float* a1  = (const float*)d_in[12];
  const float* a2  = (const float*)d_in[13];
  const float* v0  = (const float*)d_in[14];
  const float* v1  = (const float*)d_in[15];
  const float* v2  = (const float*)d_in[16];
  const float* g1  = (const float*)d_in[17];
  const float* g2  = (const float*)d_in[18];
  const float* k_k = (const float*)d_in[19];
  const float* k_a = (const float*)d_in[20];
  const float* r_k = (const float*)d_in[21];
  const float* W_r = (const float*)d_in[22];
  const float* W_k = (const float*)d_in[23];
  const float* W_v = (const float*)d_in[24];
  const float* W_o = (const float*)d_in[25];
  const float* ln_g = (const float*)d_in[26];
  const float* ln_b = (const float*)d_in[27];

  char* wsb = (char*)d_ws;
  size_t off = 0;
  auto alloc = [&](size_t bytes) -> void* {
    void* p = wsb + off;
    off += bytes;
    off = (off + 255) & ~(size_t)255;
    return p;
  };

  const size_t P32 = (size_t)BT * Cz;
  u16* xr_b = (u16*)alloc(P32 * 2);
  u16* xw_b = (u16*)alloc(P32 * 2);
  u16* xk_b = (u16*)alloc(P32 * 2);
  u16* xv_b = (u16*)alloc(P32 * 2);
  u16* xa_b = (u16*)alloc(P32 * 2);
  u16* xg_b = (u16*)alloc(P32 * 2);
  u16* Wr_b = (u16*)alloc((size_t)Cz * Cz * 2);
  u16* Wk_b = (u16*)alloc((size_t)Cz * Cz * 2);
  u16* Wv_b = (u16*)alloc((size_t)Cz * Cz * 2);
  u16* Wo_b = (u16*)alloc((size_t)Cz * Cz * 2);
  u16* w1T = (u16*)alloc((size_t)Cz * 64 * 2);    // [64,C]
  u16* w2T = (u16*)alloc((size_t)64 * Cz * 2);    // [C,64]
  u16* a1T = (u16*)alloc((size_t)Cz * 64 * 2);
  u16* a2T = (u16*)alloc((size_t)64 * Cz * 2);
  u16* v1T = (u16*)alloc((size_t)Cz * 64 * 2);
  u16* v2T = (u16*)alloc((size_t)64 * Cz * 2);
  u16* g1T = (u16*)alloc((size_t)Cz * 160 * 2);   // [160,C]
  u16* g2T = (u16*)alloc((size_t)160 * Cz * 2);   // [C,160]
  float* r32 = (float*)alloc(P32 * 4);
  float* k32 = (float*)alloc(P32 * 4);
  float* v32 = (float*)alloc(P32 * 4);
  float* hw32 = (float*)alloc((size_t)BT * 64 * 4);
  float* hv32 = (float*)alloc((size_t)BT * 64 * 4);
  float* ha32 = (float*)alloc((size_t)BT * 64 * 4);
  float* hg32 = (float*)alloc((size_t)BT * 160 * 4);
  u16* hw_b = (u16*)alloc((size_t)BT * 64 * 2);
  u16* hv_b = (u16*)alloc((size_t)BT * 64 * 2);
  u16* ha_b = (u16*)alloc((size_t)BT * 64 * 2);
  u16* hg_b = (u16*)alloc((size_t)BT * 160 * 2);
  float* dw32 = (float*)alloc(P32 * 4);
  float* dv32 = (float*)alloc(P32 * 4);
  float* da32 = (float*)alloc(P32 * 4);
  float* g32  = (float*)alloc(P32 * 4);
  float* Rs = (float*)alloc(P32 * 4);
  float* Ws = (float*)alloc(P32 * 4);
  float* Ks = (float*)alloc(P32 * 4);
  float* Vs = (float*)alloc(P32 * 4);
  float* As = (float*)alloc(P32 * 4);
  float* Bs = (float*)alloc(P32 * 4);
  float* Ys = (float*)alloc(P32 * 4);
  u16*   zo = (u16*)alloc(P32 * 2);

  int nBC = BT * Cz;
  k_mix<<<nBC / 256, 256, 0, stream>>>(x, x_r, x_w, x_k, x_v, x_a, x_g,
                                       xr_b, xw_b, xk_b, xv_b, xa_b, xg_b);

  auto cvt = [&](const float* s, u16* d, int n) {
    k_f2bf<<<(n + 255) / 256, 256, 0, stream>>>(s, d, n);
  };
  auto cvtT = [&](const float* s, u16* d, int R_, int Cc) {
    k_f2bfT<<<(R_ * Cc + 255) / 256, 256, 0, stream>>>(s, d, R_, Cc);
  };
  cvt(W_r, Wr_b, Cz * Cz); cvt(W_k, Wk_b, Cz * Cz);
  cvt(W_v, Wv_b, Cz * Cz); cvt(W_o, Wo_b, Cz * Cz);
  cvtT(w1, w1T, Cz, 64);  cvtT(w2, w2T, 64, Cz);
  cvtT(a1, a1T, Cz, 64);  cvtT(a2, a2T, 64, Cz);
  cvtT(v1, v1T, Cz, 64);  cvtT(v2, v2T, 64, Cz);
  cvtT(g1, g1T, Cz, 160); cvtT(g2, g2T, 160, Cz);

  auto gemm = [&](const u16* A_, const u16* B_, float* C_, int M, int Nn, int Kk) {
    dim3 g(Nn / 32, M / 64);
    gemm_bf16_wmma<<<g, 256, 0, stream>>>(A_, B_, C_, M, Nn, Kk);
  };

  gemm(xr_b, Wr_b, r32, BT, Cz, Cz);
  gemm(xk_b, Wk_b, k32, BT, Cz, Cz);
  gemm(xv_b, Wv_b, v32, BT, Cz, Cz);
  gemm(xw_b, w1T, hw32, BT, 64, Cz);
  gemm(xv_b, v1T, hv32, BT, 64, Cz);
  gemm(xa_b, a1T, ha32, BT, 64, Cz);
  gemm(xg_b, g1T, hg32, BT, 160, Cz);

  k_act<<<(BT * 64 + 255) / 256, 256, 0, stream>>>(hw32, hw_b, BT * 64, 1);
  k_act<<<(BT * 64 + 255) / 256, 256, 0, stream>>>(hv32, hv_b, BT * 64, 0);
  k_act<<<(BT * 64 + 255) / 256, 256, 0, stream>>>(ha32, ha_b, BT * 64, 0);
  k_act<<<(BT * 160 + 255) / 256, 256, 0, stream>>>(hg32, hg_b, BT * 160, 2);

  gemm(hw_b, w2T, dw32, BT, Cz, 64);
  gemm(hv_b, v2T, dv32, BT, Cz, 64);
  gemm(ha_b, a2T, da32, BT, Cz, 64);
  gemm(hg_b, g2T, g32, BT, Cz, 160);

  k_prep<<<nBC / 256, 256, 0, stream>>>(r32, k32, v32, dw32, da32, dv32, vfirst,
                                        w0, a0, v0, k_k, k_a,
                                        Rs, Ws, Ks, Vs, As, Bs);

  k_scan<<<Bz * Hz, 256, 0, stream>>>(Rs, Ws, Ks, Vs, As, Bs, Ys);

  k_post<<<nBC / 256, 256, 0, stream>>>(Ys, Rs, Ks, Vs, g32, ln_g, ln_b, r_k, zo);

  gemm(zo, Wo_b, (float*)d_out, BT, Cz, Cz);
}
